// Conv2d_mvm_13151189860660
// MI455X (gfx1250) — matrix operand order repaired and checked by reading, not compiled, not run on hardware
//
#include <hip/hip_runtime.h>
#include <hip/hip_bf16.h>

typedef __attribute__((ext_vector_type(16))) __bf16       v16bf;
typedef __attribute__((ext_vector_type(8)))  float        v8f;
typedef __attribute__((ext_vector_type(8)))  unsigned int v8u;

#define CIN    128
#define COUT   256
#define HW     56
#define PIX    (HW * HW)         // 3136
#define KTOT   (CIN * 9)         // 1152
#define XROWS  10                // 8 output rows + 2 halo rows
#define XCOLS  58                // 56 cols + 2 halo cols
#define XP     20                // dwords per (row,col) site: 16 chpair dwords + 4 pad (80B, b128-aligned)
#define NT     7                 // 7 x 16-pixel ntiles per wave (448 px / 4 waves / 16)

__device__ __forceinline__ unsigned short f2bf_rne(float f) {
    unsigned int u = __builtin_bit_cast(unsigned int, f);
    u += 0x7FFFu + ((u >> 16) & 1u);
    return (unsigned short)(u >> 16);
}

// One-time pre-pass: W [Cout][Cin][3][3] f32 -> Wt [Cout][rs*128 + c] bf16 (RNE)
__global__ __launch_bounds__(256) void wtrans_kernel(const float* __restrict__ W,
                                                     unsigned short* __restrict__ Wt) {
    const int idx  = blockIdx.x * 256 + threadIdx.x;  // < 256*1152
    const int cout = idx / KTOT;
    const int kp   = idx % KTOT;
    const int rs   = kp >> 7;
    const int c    = kp & 127;
    Wt[idx] = f2bf_rne(W[(size_t)cout * KTOT + c * 9 + rs]);
}

__device__ __forceinline__ v16bf load_afrag(const unsigned int* wr, int khalf) {
    v8u au;
    #pragma unroll
    for (int i = 0; i < 8; ++i)
        au[i] = wr[khalf * 8 + i];
    return __builtin_bit_cast(v16bf, au);
}

__device__ __forceinline__ v16bf load_bfrag(const unsigned int* xs, int ad) {
    const uint4 lo = *(const uint4*)&xs[ad];       // ds_load_b128
    const uint4 hi = *(const uint4*)&xs[ad + 4];   // ds_load_b128
    const v8u bu = {lo.x, lo.y, lo.z, lo.w, hi.x, hi.y, hi.z, hi.w};
    return __builtin_bit_cast(v16bf, bu);
}

__global__ __launch_bounds__(256) void conv3x3_wmma_bf16(
    const float* __restrict__ x,            // [32][128][56][56] f32
    const unsigned short* __restrict__ Wt,  // [256][1152] bf16, K = rs*128+c
    float* __restrict__ out)                // [32][256][56][56] f32
{
    __shared__ unsigned int Xs[XROWS * XCOLS * XP];   // 11600 dwords = 46.4 KB

    const int tid    = threadIdx.x;
    const int lane   = tid & 31;
    const int wave   = tid >> 5;
    const int mgroup = wave >> 2;            // 0..1 : 32-Cout half of block tile
    const int npart  = wave & 3;             // 0..3 : 112-pixel slice
    const int lm     = lane & 15;
    const int khalf  = lane >> 4;

    const int b  = blockIdx.x / 7;           // batch image
    const int rg = blockIdx.x % 7;           // output rows rg*8 .. rg*8+7
    const int coutBase = blockIdx.y * 64;

    // ---- per-lane fragment bases for the 7 ntiles ----
    int ro[NT], co[NT], fragBase[NT];
    {
        const int p0 = npart * 112 + lm;
        int rr = p0 / HW;
        int cn = p0 % HW;
        #pragma unroll
        for (int nt = 0; nt < NT; ++nt) {
            ro[nt] = rr;
            co[nt] = cn;
            fragBase[nt] = (rr * XCOLS + cn) * XP + khalf * 8;
            cn += 16;
            if (cn >= HW) { cn -= HW; rr += 1; }
        }
    }

    // ---- zero the halo pad columns (col 0 and 57) once ----
    for (int idx = tid; idx < XROWS * 2 * 16; idx += 256) {   // 320 dwords
        const int cp  = idx & 15;
        const int pos = idx >> 4;
        const int rr  = pos >> 1;
        const int col = (pos & 1) ? (XCOLS - 1) : 0;
        Xs[(rr * XCOLS + col) * XP + cp] = 0u;
    }

    v8f acc[2][NT];
    #pragma unroll
    for (int mt = 0; mt < 2; ++mt)
        #pragma unroll
        for (int nt = 0; nt < NT; ++nt)
            acc[mt][nt] = v8f{0.f,0.f,0.f,0.f,0.f,0.f,0.f,0.f};

    const unsigned int* WtU = (const unsigned int*)Wt;
    const unsigned int* wrBase0 = WtU + (size_t)(coutBase + (mgroup * 2 + 0) * 16 + lm) * (KTOT / 2);
    const unsigned int* wrBase1 = WtU + (size_t)(coutBase + (mgroup * 2 + 1) * 16 + lm) * (KTOT / 2);

    for (int cc = 0; cc < 4; ++cc) {          // 4 chunks of 32 channels
        // ---------- stage: raw x rows (with halo) for 32 channels, perm-packed bf16 pairs ----------
        for (int it = 0; it < 9; ++it) {      // 140 positions x 16 chpairs = 2240 tasks
            const int idx = it * 256 + tid;
            const int cp  = idx & 15;
            const int pos = idx >> 4;
            if (pos < 140) {
                const int rr = pos / 14;              // 0..9 -> input row rg*8+rr-1
                const int qd = pos % 14;              // image cols 4qd..4qd+3
                const int hrow = rg * 8 + rr - 1;
                const int c  = cc * 32 + 2 * cp;
                unsigned int d0 = 0u, d1 = 0u, d2 = 0u, d3 = 0u;
                if ((unsigned)hrow < (unsigned)HW) {
                    const float* pa = x + (((size_t)b * CIN + c) * HW + hrow) * HW + 4 * qd;
                    const float4 la = *(const float4*)pa;          // channel c
                    const float4 lb = *(const float4*)(pa + PIX);  // channel c+1
                    d0 = __builtin_amdgcn_perm(__builtin_bit_cast(unsigned int, lb.x),
                                               __builtin_bit_cast(unsigned int, la.x), 0x07060302u);
                    d1 = __builtin_amdgcn_perm(__builtin_bit_cast(unsigned int, lb.y),
                                               __builtin_bit_cast(unsigned int, la.y), 0x07060302u);
                    d2 = __builtin_amdgcn_perm(__builtin_bit_cast(unsigned int, lb.z),
                                               __builtin_bit_cast(unsigned int, la.z), 0x07060302u);
                    d3 = __builtin_amdgcn_perm(__builtin_bit_cast(unsigned int, lb.w),
                                               __builtin_bit_cast(unsigned int, la.w), 0x07060302u);
                }
                unsigned int* dst = &Xs[(rr * XCOLS + 4 * qd + 1) * XP + cp];
                dst[0 * XP] = d0;
                dst[1 * XP] = d1;
                dst[2 * XP] = d2;
                dst[3 * XP] = d3;
            }
        }

        // preload A fragments for tap rs=0 (global; overlaps the barrier wait)
        v16bf aCur0 = load_afrag(wrBase0 + cc * 16, khalf);
        v16bf aCur1 = load_afrag(wrBase1 + cc * 16, khalf);

        __syncthreads();

        // ---------- compute: 9 taps fully unrolled, IGroupLP-pinned pipeline ----------
        #pragma unroll
        for (int rs = 0; rs < 9; ++rs) {
            const int r    = rs / 3;                      // compile-time after unroll
            const int s    = rs % 3;
            const int roff = (r * XCOLS + s) * XP;        // folds into DS immediate offsets
            const int kgdN = ((rs < 8) ? (rs + 1) : rs) * 64 + cc * 16;

            v16bf aNxt0 = load_afrag(wrBase0 + kgdN, khalf);   // rs=8 copy is DCE'd
            v16bf aNxt1 = load_afrag(wrBase1 + kgdN, khalf);

            v16bf bf[NT];
            bf[0] = load_bfrag(Xs, fragBase[0] + roff);
            bf[1] = load_bfrag(Xs, fragBase[1] + roff);
            #pragma unroll
            for (int nt = 0; nt < NT; ++nt) {
                if (nt + 2 < NT)
                    bf[nt + 2] = load_bfrag(Xs, fragBase[nt + 2] + roff);
                acc[0][nt] = __builtin_amdgcn_wmma_f32_16x16x32_bf16(
                    false, aCur0, false, bf[nt], (short)0, acc[0][nt], false, false);
                acc[1][nt] = __builtin_amdgcn_wmma_f32_16x16x32_bf16(
                    false, aCur1, false, bf[nt], (short)0, acc[1][nt], false, false);
            }
            aCur0 = aNxt0;
            aCur1 = aNxt1;

            // --- scheduling template for this tap (DS_READ=0x100, MFMA/WMMA=0x8, VMEM_READ=0x20) ---
            __builtin_amdgcn_sched_group_barrier(0x100, 4, 0);  // bf0,bf1 in flight
            __builtin_amdgcn_sched_group_barrier(0x008, 2, 0);  // wmma(bf0) x2
            __builtin_amdgcn_sched_group_barrier(0x020, 4, 0);  // next-tap A prefetch
            #pragma unroll
            for (int g = 0; g < 4; ++g) {
                __builtin_amdgcn_sched_group_barrier(0x100, 2, 0);  // bf[g+2]
                __builtin_amdgcn_sched_group_barrier(0x008, 2, 0);  // wmma(bf[g+1]) x2
            }
            __builtin_amdgcn_sched_group_barrier(0x100, 2, 0);      // bf6
            __builtin_amdgcn_sched_group_barrier(0x008, 4, 0);      // wmma(bf5,bf6) x4
        }
        __syncthreads();
    }

    // ---------- store: tile element (M = j + 8*khalf, N per ntile) ----------
    #pragma unroll
    for (int nt = 0; nt < NT; ++nt) {
        const int hw = (rg * 8 + ro[nt]) * HW + co[nt];
        #pragma unroll
        for (int mt = 0; mt < 2; ++mt) {
            const int rowbase = coutBase + (mgroup * 2 + mt) * 16 + khalf * 8;
            #pragma unroll
            for (int j = 0; j < 8; ++j) {
                out[((size_t)b * COUT + (rowbase + j)) * PIX + hw] = acc[mt][nt][j];
            }
        }
    }
}

extern "C" void kernel_launch(void* const* d_in, const int* in_sizes, int n_in,
                              void* d_out, int out_size, void* d_ws, size_t ws_size,
                              hipStream_t stream) {
    const float* x = (const float*)d_in[0];
    const float* W = (const float*)d_in[1];
    float* out     = (float*)d_out;
    unsigned short* Wt = (unsigned short*)d_ws;   // 576 KB

    wtrans_kernel<<<dim3((COUT * KTOT) / 256), dim3(256), 0, stream>>>(W, Wt);

    // 224 pixel tiles (32 images x 7 row-groups) x 4 cout tiles, 8 waves/block
    conv3x3_wmma_bf16<<<dim3(224, 4), dim3(256), 0, stream>>>(x, Wt, out);
}
